// AudioReconModel_64407329571518
// MI455X (gfx1250) — matrix operand order repaired and checked by reading, not compiled, not run on hardware
//
#include <hip/hip_runtime.h>
#include <hip/hip_bf16.h>

typedef __bf16 bf16;
typedef __attribute__((ext_vector_type(16))) __bf16 v16bf;
typedef __attribute__((ext_vector_type(8)))  __bf16 v8bf;
typedef __attribute__((ext_vector_type(8)))  float  v8f;
typedef __attribute__((ext_vector_type(8)))  float  float8;

#define BT      8192      // B*T rows after stride-2 (8*1024)
#define TIN     2048
#define CCAT    3328
#define DCODE   256
#define NCODES  8192

#define WMMA_BF16(a, b, c) __builtin_amdgcn_wmma_f32_16x16x32_bf16( \
        false, (a), false, (b), (short)0, (c), false, false)

// ---------------- weight repack: w[oc][ic][k] fp32 -> Bmat[oc][k*C+ic] bf16 ----------------
__global__ void k_convw(const float* __restrict__ w, bf16* __restrict__ out, int N, int C) {
    int tid = blockIdx.x * 256 + threadIdx.x;
    if (tid >= N * C) return;
    int oc = tid / C, ic = tid % C;
    const float* src = w + (size_t)(oc * C + ic) * 4;
    bf16* dst = out + (size_t)oc * 4 * C;
#pragma unroll
    for (int k = 0; k < 4; ++k) dst[k * C + ic] = (bf16)src[k];
}

// ---------------- generic fp32 -> bf16 cast with strided/offset store ----------------
__global__ void k_cast(const float* __restrict__ src, bf16* __restrict__ dst,
                       long long n, int cols, int dpitch, int doff) {
    long long i = (long long)blockIdx.x * 256 + threadIdx.x;
    if (i >= n) return;
    long long r = i / cols;
    int c = (int)(i % cols);
    dst[r * dpitch + doff + c] = (bf16)src[i];
}

// ---------------- codebook: bf16 cast + exact fp32 row norms ----------------
__global__ void k_cbprep(const float* __restrict__ cb, bf16* __restrict__ cbb,
                         float* __restrict__ cnorm) {
    int row = blockIdx.x;          // 8192
    int t = threadIdx.x;           // 256
    float v = cb[(size_t)row * DCODE + t];
    cbb[(size_t)row * DCODE + t] = (bf16)v;
    float s = v * v;
#pragma unroll
    for (int m = 16; m >= 1; m >>= 1) s += __shfl_xor(s, m, 32);
    __shared__ float p[8];
    if ((t & 31) == 0) p[t >> 5] = s;
    __syncthreads();
    if (t == 0) {
        float tot = 0.f;
#pragma unroll
        for (int i = 0; i < 8; ++i) tot += p[i];
        cnorm[row] = tot;
    }
}

// ---------------- conv1d(k=4,s=2,p=1) as bf16 WMMA GEMM, writes into concat ----------------
// grid: (BT/128, N/128); block 256 (8 waves * wave32), M-tile 128, N-tile 128.
// K-loop unrolled by 2 into a ping-pong: no aF/stg register copies, one
// barrier per K-step, LDS B tile double-buffered. Ksteps is even (160/128).
__global__ __launch_bounds__(256) void k_conv(const float* __restrict__ x,
                                              const bf16* __restrict__ Bmat,
                                              const float* __restrict__ bias,
                                              bf16* __restrict__ outcat,
                                              int C, int coloff) {
    __shared__ __align__(64) bf16 Blds[2][128 * 32];
    const int lane = threadIdx.x & 31;
    const int wave = threadIdx.x >> 5;
    const int rowbase = blockIdx.x * 128 + wave * 16;
    const int nblock = blockIdx.y * 128;
    const int row = rowbase + (lane & 15);
    const int b = row >> 10;
    const int t = row & 1023;
    const int khalf8 = (lane >> 4) * 8;
    const int Ktot = 4 * C;
    const int Ksteps = Ktot / 32;
    const int fragoff = (lane & 15) * 32 + khalf8 * 2;   // + nt*512
    const int brow = threadIdx.x >> 1;                   // cooperative B row
    const int bhalf = (threadIdx.x & 1) * 16;            // 16 bf16 per half

    v8f acc[8];
#pragma unroll
    for (int nt = 0; nt < 8; ++nt)
#pragma unroll
        for (int j = 0; j < 8; ++j) acc[nt][j] = 0.f;

    const bf16* gB = Bmat + (size_t)(nblock + brow) * Ktot + bhalf;

    auto loadA = [&](int kk, int icb) -> v16bf {
        v16bf a;
        int tt = 2 * t - 1 + kk;
        if (tt >= 0 && tt < TIN) {
            const float* ax = x + ((size_t)b * TIN + tt) * C + icb + khalf8 * 2;
            float8 lo = *(const float8*)(ax);
            float8 hi = *(const float8*)(ax + 8);
#pragma unroll
            for (int i = 0; i < 8; ++i) { a[i] = (bf16)lo[i]; a[i + 8] = (bf16)hi[i]; }
        } else {
#pragma unroll
            for (int i = 0; i < 16; ++i) a[i] = (bf16)0.f;
        }
        return a;
    };
    auto compute = [&](const bf16* Sb, v16bf a) {
        v16bf f0 = *(const v16bf*)(Sb + 0 * 512);
        v16bf f1 = *(const v16bf*)(Sb + 1 * 512);
        v16bf f2 = *(const v16bf*)(Sb + 2 * 512);
        v16bf f3 = *(const v16bf*)(Sb + 3 * 512);
        v16bf f4 = *(const v16bf*)(Sb + 4 * 512);
        v16bf f5 = *(const v16bf*)(Sb + 5 * 512);
        v16bf f6 = *(const v16bf*)(Sb + 6 * 512);
        v16bf f7 = *(const v16bf*)(Sb + 7 * 512);
        acc[0] = WMMA_BF16(a, f0, acc[0]);
        acc[1] = WMMA_BF16(a, f1, acc[1]);
        acc[2] = WMMA_BF16(a, f2, acc[2]);
        acc[3] = WMMA_BF16(a, f3, acc[3]);
        acc[4] = WMMA_BF16(a, f4, acc[4]);
        acc[5] = WMMA_BF16(a, f5, acc[5]);
        acc[6] = WMMA_BF16(a, f6, acc[6]);
        acc[7] = WMMA_BF16(a, f7, acc[7]);
    };

    // ---- prologue: stage tile 0 into buffer 0, A frag 0 ----
    {
        v16bf stg0 = *(const v16bf*)(gB);
        *(v16bf*)(&Blds[0][brow * 32 + bhalf]) = stg0;
    }
    v16bf aF0 = loadA(0, 0);
    int kn = 0, icn = 32;               // (k, icbase) of the NEXT tile to prefetch
    if (icn == C) { icn = 0; kn = 1; }

    for (int ks = 0; ks < Ksteps; ks += 2) {
        // ---- even sub-iteration: compute buf0/aF0, prefetch ks+1 into buf1/aF1 ----
        v16bf stg1 = *(const v16bf*)(gB + (ks + 1) * 32);
        v16bf aF1 = loadA(kn, icn);
        icn += 32;
        if (icn == C) { icn = 0; ++kn; }
        __syncthreads();
        compute(&Blds[0][fragoff], aF0);
        *(v16bf*)(&Blds[1][brow * 32 + bhalf]) = stg1;

        // ---- odd sub-iteration: compute buf1/aF1, prefetch ks+2 into buf0/aF0 ----
        const bool have2 = (ks + 2) < Ksteps;
        v16bf stg0;
        if (have2) {
            stg0 = *(const v16bf*)(gB + (ks + 2) * 32);
            aF0 = loadA(kn, icn);
            icn += 32;
            if (icn == C) { icn = 0; ++kn; }
        }
        __syncthreads();
        compute(&Blds[1][fragoff], aF1);
        if (have2) *(v16bf*)(&Blds[0][brow * 32 + bhalf]) = stg0;
    }

    const int srow = rowbase + (lane >> 4) * 8;
#pragma unroll
    for (int nt = 0; nt < 8; ++nt) {
        int col = nblock + nt * 16 + (lane & 15);
        float bv = bias[col];
#pragma unroll
        for (int j = 0; j < 8; ++j) {
            int r = srow + j;
            outcat[(size_t)r * CCAT + coloff + col] = (bf16)(acc[nt][j] + bv);
        }
    }
}

// ---------------- projection GEMM: z_e = concat @ proj_w^T + b ----------------
// grid: (BT/128, 2); block 256. Same ping-pong pipeline (Ksteps=104, even).
__global__ __launch_bounds__(256) void k_proj(const bf16* __restrict__ Acat,
                                              const bf16* __restrict__ Bw,
                                              const float* __restrict__ bias,
                                              float* __restrict__ zef,
                                              bf16* __restrict__ zeb) {
    __shared__ __align__(64) bf16 Blds[2][128 * 32];
    const int lane = threadIdx.x & 31;
    const int wave = threadIdx.x >> 5;
    const int rowbase = blockIdx.x * 128 + wave * 16;
    const int nblock = blockIdx.y * 128;
    const int arow = rowbase + (lane & 15);
    const int khalf8 = (lane >> 4) * 8;
    const int Ksteps = CCAT / 32;
    const int fragoff = (lane & 15) * 32 + khalf8 * 2;
    const int brow = threadIdx.x >> 1;
    const int bhalf = (threadIdx.x & 1) * 16;

    v8f acc[8];
#pragma unroll
    for (int nt = 0; nt < 8; ++nt)
#pragma unroll
        for (int j = 0; j < 8; ++j) acc[nt][j] = 0.f;

    const bf16* gB = Bw + (size_t)(nblock + brow) * CCAT + bhalf;
    const bf16* gA = Acat + (size_t)arow * CCAT + khalf8 * 2;

    auto loadA = [&](int ksIdx) -> v16bf {
        v16bf a;
        const bf16* ap = gA + ksIdx * 32;
        v8bf lo = *(const v8bf*)(ap);
        v8bf hi = *(const v8bf*)(ap + 8);
#pragma unroll
        for (int i = 0; i < 8; ++i) { a[i] = lo[i]; a[i + 8] = hi[i]; }
        return a;
    };
    auto compute = [&](const bf16* Sb, v16bf a) {
        v16bf f0 = *(const v16bf*)(Sb + 0 * 512);
        v16bf f1 = *(const v16bf*)(Sb + 1 * 512);
        v16bf f2 = *(const v16bf*)(Sb + 2 * 512);
        v16bf f3 = *(const v16bf*)(Sb + 3 * 512);
        v16bf f4 = *(const v16bf*)(Sb + 4 * 512);
        v16bf f5 = *(const v16bf*)(Sb + 5 * 512);
        v16bf f6 = *(const v16bf*)(Sb + 6 * 512);
        v16bf f7 = *(const v16bf*)(Sb + 7 * 512);
        acc[0] = WMMA_BF16(a, f0, acc[0]);
        acc[1] = WMMA_BF16(a, f1, acc[1]);
        acc[2] = WMMA_BF16(a, f2, acc[2]);
        acc[3] = WMMA_BF16(a, f3, acc[3]);
        acc[4] = WMMA_BF16(a, f4, acc[4]);
        acc[5] = WMMA_BF16(a, f5, acc[5]);
        acc[6] = WMMA_BF16(a, f6, acc[6]);
        acc[7] = WMMA_BF16(a, f7, acc[7]);
    };

    {
        v16bf stg0 = *(const v16bf*)(gB);
        *(v16bf*)(&Blds[0][brow * 32 + bhalf]) = stg0;
    }
    v16bf aF0 = loadA(0);

    for (int ks = 0; ks < Ksteps; ks += 2) {
        v16bf stg1 = *(const v16bf*)(gB + (ks + 1) * 32);
        v16bf aF1 = loadA(ks + 1);
        __syncthreads();
        compute(&Blds[0][fragoff], aF0);
        *(v16bf*)(&Blds[1][brow * 32 + bhalf]) = stg1;

        const bool have2 = (ks + 2) < Ksteps;
        v16bf stg0;
        if (have2) {
            stg0 = *(const v16bf*)(gB + (ks + 2) * 32);
            aF0 = loadA(ks + 2);
        }
        __syncthreads();
        compute(&Blds[1][fragoff], aF1);
        if (have2) *(v16bf*)(&Blds[0][brow * 32 + bhalf]) = stg0;
    }

    const int srow = rowbase + (lane >> 4) * 8;
#pragma unroll
    for (int nt = 0; nt < 8; ++nt) {
        int col = nblock + nt * 16 + (lane & 15);
        float bv = bias[col];
#pragma unroll
        for (int j = 0; j < 8; ++j) {
            float v = acc[nt][j] + bv;
            size_t idx = (size_t)(srow + j) * DCODE + col;
            zef[idx] = v;
            zeb[idx] = (bf16)v;
        }
    }
}

// ---------------- VQ: argmin_c (||c||^2 - 2 z.c) via WMMA, LDS-staged codebook ----------------
// grid: BT/128; block 256. 64-code tile (32 KB LDS), register-staged prefetch.
__global__ __launch_bounds__(256) void k_vq(const bf16* __restrict__ zeb,
                                            const bf16* __restrict__ cbb,
                                            const float* __restrict__ cnorm,
                                            int* __restrict__ codes,
                                            float* __restrict__ codes_f) {
    __shared__ __align__(64) bf16 Clds[64 * DCODE];    // 32 KB code tile
    const int lane = threadIdx.x & 31;
    const int wave = threadIdx.x >> 5;
    const int rowbase = blockIdx.x * 128 + wave * 16;
    const int arow = rowbase + (lane & 15);
    const int khalf8 = (lane >> 4) * 8;

    // keep this wave's 16 z_e rows resident in A fragments (K=256)
    v16bf aF[8];
#pragma unroll
    for (int ks = 0; ks < 8; ++ks) {
        const bf16* ap = zeb + (size_t)arow * DCODE + ks * 32 + khalf8 * 2;
        v8bf lo = *(const v8bf*)(ap);
        v8bf hi = *(const v8bf*)(ap + 8);
#pragma unroll
        for (int i = 0; i < 8; ++i) { aF[ks][i] = lo[i]; aF[ks][i + 8] = hi[i]; }
    }

    float best[8];
    int bidx[8];
#pragma unroll
    for (int j = 0; j < 8; ++j) { best[j] = 3.4e38f; bidx[j] = 0x7fffffff; }

    // stage tile 0 into LDS (64 codes x 256 dims = 2048 x v8bf)
    v8bf st[8];
    {
        const v8bf* g = (const v8bf*)(cbb);
#pragma unroll
        for (int i = 0; i < 8; ++i) st[i] = g[i * 256 + threadIdx.x];
        v8bf* s = (v8bf*)Clds;
#pragma unroll
        for (int i = 0; i < 8; ++i) s[i * 256 + threadIdx.x] = st[i];
    }

    for (int ct = 0; ct < NCODES / 64; ++ct) {
        const bool have = (ct + 1) < NCODES / 64;
        if (have) {
            const v8bf* g = (const v8bf*)(cbb + (size_t)(ct + 1) * 64 * DCODE);
#pragma unroll
            for (int i = 0; i < 8; ++i) st[i] = g[i * 256 + threadIdx.x];
        }
        __syncthreads();   // LDS stores of this tile are visible

#pragma unroll
        for (int nt = 0; nt < 4; ++nt) {
            v8f acc;
#pragma unroll
            for (int j = 0; j < 8; ++j) acc[j] = 0.f;
            const bf16* Sb = &Clds[(nt * 16 + (lane & 15)) * DCODE + khalf8 * 2];
#pragma unroll
            for (int g = 0; g < 2; ++g) {
                v16bf b0 = *(const v16bf*)(Sb + (g * 4 + 0) * 32);
                v16bf b1 = *(const v16bf*)(Sb + (g * 4 + 1) * 32);
                v16bf b2 = *(const v16bf*)(Sb + (g * 4 + 2) * 32);
                v16bf b3 = *(const v16bf*)(Sb + (g * 4 + 3) * 32);
                acc = WMMA_BF16(aF[g * 4 + 0], b0, acc);
                acc = WMMA_BF16(aF[g * 4 + 1], b1, acc);
                acc = WMMA_BF16(aF[g * 4 + 2], b2, acc);
                acc = WMMA_BF16(aF[g * 4 + 3], b3, acc);
            }
            int code = ct * 64 + nt * 16 + (lane & 15);
            float cn = cnorm[code];
#pragma unroll
            for (int j = 0; j < 8; ++j) {
                float d = cn - 2.f * acc[j];
                if (d < best[j]) { best[j] = d; bidx[j] = code; }  // ascending code -> first-min
            }
        }
        __syncthreads();   // all reads of this tile complete
        if (have) {
            v8bf* s = (v8bf*)Clds;
#pragma unroll
            for (int i = 0; i < 8; ++i) s[i * 256 + threadIdx.x] = st[i];
        }
    }

    // reduce across the 16 lanes of each half (C layout: lanes<16 rows j, lanes>=16 rows j+8)
#pragma unroll
    for (int j = 0; j < 8; ++j) {
#pragma unroll
        for (int m = 1; m < 16; m <<= 1) {
            float od = __shfl_xor(best[j], m, 32);
            int oi = __shfl_xor(bidx[j], m, 32);
            if (od < best[j] || (od == best[j] && oi < bidx[j])) { best[j] = od; bidx[j] = oi; }
        }
    }
    if (lane == 0 || lane == 16) {
        int rb = rowbase + (lane >> 4) * 8;
#pragma unroll
        for (int j = 0; j < 8; ++j) {
            codes[rb + j] = bidx[j];
            codes_f[rb + j] = (float)bidx[j];
        }
    }
}

// ---------------- gather z_q (exact fp32) + per-row commit-loss partials ----------------
__global__ void k_gather(const int* __restrict__ codes, const float* __restrict__ cb,
                         const float* __restrict__ zef, float* __restrict__ zq,
                         float* __restrict__ partials) {
    int row = blockIdx.x;      // 8192
    int t = threadIdx.x;       // 256
    int code = codes[row];
    float q = cb[(size_t)code * DCODE + t];
    zq[(size_t)row * DCODE + t] = q;
    float d = zef[(size_t)row * DCODE + t] - q;
    float s = d * d;
#pragma unroll
    for (int m = 16; m >= 1; m >>= 1) s += __shfl_xor(s, m, 32);
    __shared__ float p[8];
    if ((t & 31) == 0) p[t >> 5] = s;
    __syncthreads();
    if (t == 0) {
        float tot = 0.f;
#pragma unroll
        for (int i = 0; i < 8; ++i) tot += p[i];
        partials[row] = tot;
    }
}

__global__ void k_final(const float* __restrict__ partials, float* __restrict__ out) {
    if (threadIdx.x == 0) {
        float tot = 0.f;
        for (int i = 0; i < BT; ++i) tot += partials[i];   // fixed order -> deterministic
        out[0] = tot / (float)(BT * DCODE);
    }
}

extern "C" void kernel_launch(void* const* d_in, const int* in_sizes, int n_in,
                              void* d_out, int out_size, void* d_ws, size_t ws_size,
                              hipStream_t stream) {
    (void)in_sizes; (void)n_in; (void)out_size; (void)ws_size;
    const float* whisper = (const float*)d_in[0];
    const float* wavlm   = (const float*)d_in[1];
    const float* muq     = (const float*)d_in[2];
    const float* wcw     = (const float*)d_in[3];
    const float* wcb     = (const float*)d_in[4];
    const float* wlw     = (const float*)d_in[5];
    const float* wlb     = (const float*)d_in[6];
    const float* pw      = (const float*)d_in[7];
    const float* pb      = (const float*)d_in[8];
    const float* cb      = (const float*)d_in[9];

    float* out = (float*)d_out;
    float* zq_out    = out;                      // 8192*256
    float* codes_out = out + (size_t)BT * DCODE; // 8192
    float* loss_out  = codes_out + BT;           // 1

    char* ws = (char*)d_ws;
    auto alloc = [&](size_t bytes) -> char* {
        char* p = ws;
        ws += (bytes + 255) & ~(size_t)255;
        return p;
    };
    bf16*  BmatW   = (bf16*) alloc((size_t)1280 * 5120 * 2);
    bf16*  BmatWL  = (bf16*) alloc((size_t)1024 * 4096 * 2);
    bf16*  projWb  = (bf16*) alloc((size_t)DCODE * CCAT * 2);
    bf16*  cbb     = (bf16*) alloc((size_t)NCODES * DCODE * 2);
    float* cnorm   = (float*)alloc((size_t)NCODES * 4);
    bf16*  concat  = (bf16*) alloc((size_t)BT * CCAT * 2);
    float* zef     = (float*)alloc((size_t)BT * DCODE * 4);
    bf16*  zeb     = (bf16*) alloc((size_t)BT * DCODE * 2);
    int*   codes   = (int*)  alloc((size_t)BT * 4);
    float* partial = (float*)alloc((size_t)BT * 4);

    // weight / codebook preparation
    k_convw<<<(1280 * 1280 + 255) / 256, 256, 0, stream>>>(wcw, BmatW, 1280, 1280);
    k_convw<<<(1024 * 1024 + 255) / 256, 256, 0, stream>>>(wlw, BmatWL, 1024, 1024);
    k_cast<<<(int)(((long long)DCODE * CCAT + 255) / 256), 256, 0, stream>>>(
        pw, projWb, (long long)DCODE * CCAT, CCAT, CCAT, 0);
    k_cast<<<(int)(((long long)BT * 1024 + 255) / 256), 256, 0, stream>>>(
        muq, concat, (long long)BT * 1024, 1024, CCAT, 2304);
    k_cbprep<<<NCODES, 256, 0, stream>>>(cb, cbb, cnorm);

    // convs (write bf16 into concat columns)
    dim3 gw(BT / 128, 1280 / 128), gwl(BT / 128, 1024 / 128);
    k_conv<<<gw,  256, 0, stream>>>(whisper, BmatW,  wcb, concat, 1280, 0);
    k_conv<<<gwl, 256, 0, stream>>>(wavlm,   BmatWL, wlb, concat, 1024, 1280);

    // projection, VQ, gather, loss
    dim3 gp(BT / 128, DCODE / 128);
    k_proj<<<gp, 256, 0, stream>>>(concat, projWb, pb, zef, zeb);
    k_vq<<<BT / 128, 256, 0, stream>>>(zeb, cbb, cnorm, codes, codes_out);
    k_gather<<<BT, 256, 0, stream>>>(codes, cb, zef, zq_out, partial);
    k_final<<<1, 32, 0, stream>>>(partial, loss_out);
}
